// FEDformerLayer_32057635897486
// MI455X (gfx1250) — matrix operand order repaired and checked by reading, not compiled, not run on hardware
//
#include <hip/hip_runtime.h>
#include <hip/hip_bf16.h>

// Problem constants (from reference)
#define BB   8
#define TT   4096
#define DD   512
#define HH   8
#define EE   64
#define MMM  4
#define NHID 2048   // CM*D

typedef __attribute__((ext_vector_type(16))) __bf16 v16bf;
typedef __attribute__((ext_vector_type(8)))  float  v8f;

__device__ __forceinline__ unsigned short f2bf(float f) {
  unsigned u = __float_as_uint(f);
  u += 0x7FFFu + ((u >> 16) & 1u);   // round-to-nearest-even
  return (unsigned short)(u >> 16);
}

union Frag16 { uint4 u[2]; v16bf v; };
static_assert(sizeof(Frag16) == 32, "frag size");

// LDS tile geometry: 256x64 A tile + 64x64 B tile, row stride padded to 144 B
#define LDS_STRIDE 144
#define A_ROWS 256
#define A_TILE_BYTES (A_ROWS * LDS_STRIDE)             // 36864
#define B_TILE_BYTES (64 * LDS_STRIDE)                 // 9216
#define BUF_BYTES    (A_TILE_BYTES + B_TILE_BYTES)     // 46080

// One 16B chunk per lane per instruction; ASYNCcnt-tracked async copy to LDS.
__device__ __forceinline__ void async_b128_to_lds(unsigned lds_off,
                                                  const unsigned short* gptr) {
  unsigned long long ga = (unsigned long long)gptr;
  asm volatile("global_load_async_to_lds_b128 %0, %1, off"
               :: "v"(lds_off), "v"(ga)
               : "memory");
}

__device__ __forceinline__ void wait_asynccnt0() {
  asm volatile("s_wait_asynccnt 0x0" ::: "memory");
}

// Stage one 64-deep K block of the A (256 rows) and B (64 rows) tiles.
// A: 256 rows x 8 chunks = 2048 chunks -> 8 instructions x 256 threads
// B:  64 rows x 8 chunks =  512 chunks -> 2 instructions x 256 threads
__device__ __forceinline__ void issue_tile_copies(
    const unsigned short* __restrict__ A, const unsigned short* __restrict__ Bw,
    int K, int rowBase0, int colBase, int kb,
    unsigned ldsA, unsigned ldsB, int tid)
{
  #pragma unroll
  for (int i = 0; i < 8; ++i) {
    int id = tid + i * 256;
    int r = id >> 3, c = id & 7;
    async_b128_to_lds(ldsA + (unsigned)(r * LDS_STRIDE + c * 16),
                      A + (size_t)(rowBase0 + r) * K + kb + c * 8);
  }
  #pragma unroll
  for (int i = 0; i < 2; ++i) {
    int id = tid + i * 256;
    int r = id >> 3, c = id & 7;
    async_b128_to_lds(ldsB + (unsigned)(r * LDS_STRIDE + c * 16),
                      Bw + (size_t)(colBase + r) * K + kb + c * 8);
  }
}

// ---------------------------------------------------------------------------
// WMMA GEMM: C[M,N] = A[M,K] * Bw[N,K]^T  (bf16 in, f32 accumulate)
// Double-buffered async global->LDS staging, K-step 64.
// MODE 0: += bias[col], f32 out      (projection)
// MODE 1: relu, bf16 out             (FFN up)
// MODE 2: += resid[row,col], f32 out (FFN down + residual)
// Block: 256 threads = 8 waves; tile 256 rows x 64 cols;
// each wave: 32 rows x 64 cols via 8 f32 accumulators (2 A frags x 4 B frags).
// ---------------------------------------------------------------------------
template<int MODE>
__global__ __launch_bounds__(256)
void gemm_wmma_bf16(const unsigned short* __restrict__ A,
                    const unsigned short* __restrict__ Bw,
                    const float* __restrict__ bias,
                    const float* __restrict__ resid,
                    float* __restrict__ Cf,
                    unsigned short* __restrict__ Cb,
                    int M, int N, int K)
{
  __shared__ unsigned char smem[2 * BUF_BYTES];   // 90 KB (of 320 KB / WGP)

  const int tid  = threadIdx.x;
  const int lane = tid & 31;
  const int wave = tid >> 5;
  const int half = lane >> 4;     // 0: lanes 0-15, 1: lanes 16-31
  const int l15  = lane & 15;

  const int rowBase0 = blockIdx.y * A_ROWS;
  const int colBase  = blockIdx.x * 64;

  // Flat LDS addresses have the aperture only in bits [63:32]; low 32 bits
  // are the wave-relative LDS byte offset required by async-to-LDS VDST.
  const unsigned ldsBase = (unsigned)(unsigned long long)(&smem[0]);

  const v8f zero = {0.f,0.f,0.f,0.f,0.f,0.f,0.f,0.f};
  v8f acc0[4] = {zero, zero, zero, zero};   // rows [wave*32 .. +16)
  v8f acc1[4] = {zero, zero, zero, zero};   // rows [wave*32+16 .. +32)

  // Prologue: stage K block 0 into buffer 0.
  issue_tile_copies(A, Bw, K, rowBase0, colBase, 0,
                    ldsBase, ldsBase + A_TILE_BYTES, tid);

  const int nIter = K >> 6;
  for (int it = 0; it < nIter; ++it) {
    wait_asynccnt0();       // our copies for this iteration landed
    __syncthreads();        // everyone's copies landed; prev reads done

    if (it + 1 < nIter) {
      const unsigned nb = (unsigned)((it + 1) & 1) * BUF_BYTES;
      issue_tile_copies(A, Bw, K, rowBase0, colBase, (it + 1) << 6,
                        ldsBase + nb, ldsBase + nb + A_TILE_BYTES, tid);
    }

    const unsigned char* sa = smem + (unsigned)(it & 1) * BUF_BYTES;
    const unsigned char* sb = sa + A_TILE_BYTES;
    const int ar0 = (wave * 32 + l15) * LDS_STRIDE;
    const int ar1 = ar0 + 16 * LDS_STRIDE;

    #pragma unroll
    for (int sub = 0; sub < 64; sub += 32) {
      // Materialize ALL fragments first so the backend can emit one ds
      // clause + single wait, then issue the 8 WMMAs back-to-back.
      // A fragment per ISA 16-bit A 16x32 layout:
      //   lanes 0-15 : v0..v3 = K 0..7,  v4..v7 = K 16..23
      //   lanes16-31 : v0..v3 = K 8..15, v4..v7 = K 24..31
      Frag16 a0, a1, bf[4];
      a0.u[0] = *reinterpret_cast<const uint4*>(sa + ar0 + (sub + half * 8) * 2);
      a0.u[1] = *reinterpret_cast<const uint4*>(sa + ar0 + (sub + 16 + half * 8) * 2);
      a1.u[0] = *reinterpret_cast<const uint4*>(sa + ar1 + (sub + half * 8) * 2);
      a1.u[1] = *reinterpret_cast<const uint4*>(sa + ar1 + (sub + 16 + half * 8) * 2);
      #pragma unroll
      for (int j = 0; j < 4; ++j) {
        // B fragment per ISA 16-bit B 32x16 layout:
        //   lanes 0-15 (N=l15): v0..v3 = K 0..7, v4..v7 = K 16..23 ; lanes 16-31: K 8..15, K 24..31
        const unsigned char* pb =
            sb + (j * 16 + l15) * LDS_STRIDE + (sub + half * 8) * 2;
        bf[j].u[0] = reinterpret_cast<const uint4*>(pb)[0];
        bf[j].u[1] = reinterpret_cast<const uint4*>(pb)[2];
      }
      #pragma unroll
      for (int j = 0; j < 4; ++j) {
        acc0[j] = __builtin_amdgcn_wmma_f32_16x16x32_bf16(
            false, a0.v, false, bf[j].v, (short)0, acc0[j], false, false);
        acc1[j] = __builtin_amdgcn_wmma_f32_16x16x32_bf16(
            false, a1.v, false, bf[j].v, (short)0, acc1[j], false, false);
      }
    }
  }

  // C/D layout: lanes 0-15 -> N=l15, VGPR r -> M=r ; lanes 16-31 -> M=8+r
  #pragma unroll
  for (int i = 0; i < 2; ++i) {
    const int rowBase = rowBase0 + wave * 32 + i * 16;
    #pragma unroll
    for (int j = 0; j < 4; ++j) {
      const int col = colBase + j * 16 + l15;
      float bv = 0.f;
      if (MODE == 0) bv = bias[col];
      #pragma unroll
      for (int r = 0; r < 8; ++r) {
        const int row = rowBase + half * 8 + r;
        const size_t o = (size_t)row * (size_t)N + col;
        float v = (i == 0) ? acc0[j][r] : acc1[j][r];
        if (MODE == 0) {
          Cf[o] = v + bv;
        } else if (MODE == 1) {
          Cb[o] = f2bf(v > 0.f ? v : 0.f);
        } else {
          Cf[o] = v + resid[o];
        }
      }
    }
  }
}

// ---------------------------------------------------------------------------
// Conversion helpers
// ---------------------------------------------------------------------------
__global__ void k_f32_to_bf16(const float* __restrict__ in,
                              unsigned short* __restrict__ out, int n)
{
  int i = blockIdx.x * blockDim.x + threadIdx.x;
  if (i < n) out[i] = f2bf(in[i]);
}

// in[R][C] (f32) -> out[C][R] (bf16)
__global__ void k_transpose_bf16(const float* __restrict__ in,
                                 unsigned short* __restrict__ out, int R, int C)
{
  int i = blockIdx.x * blockDim.x + threadIdx.x;
  if (i < R * C) {
    int r = i / C, c = i % C;
    out[(size_t)c * R + r] = f2bf(in[i]);
  }
}

// ---------------------------------------------------------------------------
// 4-mode DFT: X[k] = sum_t q[b,t,d] * exp(-2pi i k t / T) for k in mode_index.
// One block (256 thr = 8 waves) per series (b,d); shuffle + LDS reduction.
// ---------------------------------------------------------------------------
__global__ __launch_bounds__(256)
void k_dft_modes(const float* __restrict__ q, const int* __restrict__ mode_index,
                 float* __restrict__ Xre, float* __restrict__ Xim)
{
  const int s = blockIdx.x;           // 0 .. B*D-1
  const int b = s / DD, d = s % DD;
  const float* base = q + (size_t)b * TT * DD + d;

  int km[MMM];
  #pragma unroll
  for (int m = 0; m < MMM; ++m) km[m] = mode_index[m];

  const float w0 = 6.283185307179586f / (float)TT;
  float re[MMM] = {0.f, 0.f, 0.f, 0.f};
  float im[MMM] = {0.f, 0.f, 0.f, 0.f};

  for (int t = threadIdx.x; t < TT; t += 256) {
    float xv = base[(size_t)t * DD];
    #pragma unroll
    for (int m = 0; m < MMM; ++m) {
      float sn, cs;
      __sincosf(w0 * (float)km[m] * (float)t, &sn, &cs);
      re[m] += xv * cs;
      im[m] -= xv * sn;
    }
  }

  // wave32 reduction
  #pragma unroll
  for (int m = 0; m < MMM; ++m) {
    for (int o = 16; o > 0; o >>= 1) {
      re[m] += __shfl_xor(re[m], o, 32);
      im[m] += __shfl_xor(im[m], o, 32);
    }
  }

  __shared__ float sm[8][8];
  const int lane = threadIdx.x & 31, wv = threadIdx.x >> 5;
  if (lane == 0) {
    #pragma unroll
    for (int m = 0; m < MMM; ++m) { sm[wv][m] = re[m]; sm[wv][m + 4] = im[m]; }
  }
  __syncthreads();
  if (threadIdx.x < 8) {
    float sum = 0.f;
    #pragma unroll
    for (int w = 0; w < 8; ++w) sum += sm[w][threadIdx.x];
    int m = threadIdx.x & 3;
    size_t o = ((size_t)b * DD + d) * MMM + m;
    if (threadIdx.x < 4) Xre[o] = sum; else Xim[o] = sum;
  }
}

// ---------------------------------------------------------------------------
// Per-mode complex mixing: om[b,h,o,m] = sum_i X[b,h,i,m] * (wr + i wi)[h,i,o,m]
// ---------------------------------------------------------------------------
__global__ void k_mix(const float* __restrict__ Xre, const float* __restrict__ Xim,
                      const float* __restrict__ wr,  const float* __restrict__ wi,
                      float* __restrict__ Ore, float* __restrict__ Oim)
{
  int idx = blockIdx.x * blockDim.x + threadIdx.x;
  if (idx >= BB * HH * EE * MMM) return;
  int m = idx % MMM;
  int o = (idx / MMM) % EE;
  int h = (idx / (MMM * EE)) % HH;
  int b = idx / (MMM * EE * HH);

  float ar = 0.f, ai = 0.f;
  for (int i = 0; i < EE; ++i) {
    size_t xo = ((size_t)b * DD + h * EE + i) * MMM + m;
    float xr = Xre[xo], xi = Xim[xo];
    size_t wo = (((size_t)(h * EE + i)) * EE + o) * MMM + m;
    float wrr = wr[wo], wii = wi[wo];
    ar += xr * wrr - xi * wii;
    ai += xr * wii + xi * wrr;
  }
  size_t oo = ((size_t)b * DD + h * EE + o) * MMM + m;
  Ore[oo] = ar;
  Oim[oo] = ai;
}

// ---------------------------------------------------------------------------
// Sparse irfft + residual:
// x1 = x + (1/T) * sum_m wgt_m * (Re(om)cos - Im(om)sin), wgt=1 for k==0/Nyq
// (c2r drops the imaginary part of bin 0 / Nyquist). Writes f32 and bf16.
// ---------------------------------------------------------------------------
__global__ __launch_bounds__(256)
void k_synth_residual(const float* __restrict__ x,
                      const float* __restrict__ Ore, const float* __restrict__ Oim,
                      const int* __restrict__ mode_index,
                      float* __restrict__ x1f, unsigned short* __restrict__ x1b)
{
  size_t n = (size_t)blockIdx.x * 256 + threadIdx.x;   // over B*T*D
  int d = (int)(n % DD);
  size_t bt = n / DD;
  int t = (int)(bt % TT);
  int b = (int)(bt / TT);
  size_t base = ((size_t)b * DD + d) * MMM;

  const float w0 = 6.283185307179586f / (float)TT;
  float sum = 0.f;
  #pragma unroll
  for (int m = 0; m < MMM; ++m) {
    int k = mode_index[m];
    float orv = Ore[base + m], oiv = Oim[base + m];
    float sn, cs;
    __sincosf(w0 * (float)k * (float)t, &sn, &cs);
    bool realonly = (k == 0) || (2 * k == TT);
    float wgt = realonly ? 1.f : 2.f;
    float oim = realonly ? 0.f : oiv;
    sum += wgt * (orv * cs - oim * sn);
  }
  sum *= (1.f / (float)TT);
  float v = x[n] + sum;
  x1f[n] = v;
  x1b[n] = f2bf(v);
}

// ---------------------------------------------------------------------------
// Launch
// ---------------------------------------------------------------------------
extern "C" void kernel_launch(void* const* d_in, const int* in_sizes, int n_in,
                              void* d_out, int out_size, void* d_ws, size_t ws_size,
                              hipStream_t stream)
{
  (void)in_sizes; (void)n_in; (void)out_size; (void)ws_size;

  const float* x      = (const float*)d_in[0];
  const float* Wp     = (const float*)d_in[1];
  const float* bp     = (const float*)d_in[2];
  const float* w_real = (const float*)d_in[3];
  const float* w_imag = (const float*)d_in[4];
  const float* W1     = (const float*)d_in[5];
  const float* W2     = (const float*)d_in[6];
  const int*   midx   = (const int*)d_in[7];
  float* out = (float*)d_out;
  char*  ws  = (char*)d_ws;

  const size_t BTD = (size_t)BB * TT * DD;    // 16,777,216
  const int    Mrows = BB * TT;               // 32768

  // Workspace layout (regions 0..128MB reused: xbf+q die before y is written)
  unsigned short* xbf  = (unsigned short*)(ws);                           // 32 MB
  float*          qf   = (float*)(ws + ((size_t)32 << 20));               // 64 MB
  unsigned short* ybf  = (unsigned short*)(ws);                           // 128 MB (reuse)
  float*          x1f  = (float*)(ws + ((size_t)128 << 20));              // 64 MB
  unsigned short* x1bf = (unsigned short*)(ws + ((size_t)192 << 20));     // 32 MB
  unsigned short* wptb = (unsigned short*)(ws + ((size_t)224 << 20));     // 512 KB
  unsigned short* w1b  = (unsigned short*)(ws + ((size_t)224 << 20) + ((size_t)1 << 20)); // 2 MB
  unsigned short* w2b  = (unsigned short*)(ws + ((size_t)224 << 20) + ((size_t)3 << 20)); // 2 MB
  float* Xre = (float*)(ws + ((size_t)224 << 20) + ((size_t)5 << 20));
  float* Xim = Xre + (size_t)BB * DD * MMM;
  float* Ore = Xim + (size_t)BB * DD * MMM;
  float* Oim = Ore + (size_t)BB * DD * MMM;

  // 1) bf16 conversions
  k_f32_to_bf16<<<(unsigned)(BTD / 256), 256, 0, stream>>>(x, xbf, (int)BTD);
  k_transpose_bf16<<<(DD * DD + 255) / 256, 256, 0, stream>>>(Wp, wptb, DD, DD);
  k_f32_to_bf16<<<(NHID * DD) / 256, 256, 0, stream>>>(W1, w1b, NHID * DD);
  k_f32_to_bf16<<<(DD * NHID) / 256, 256, 0, stream>>>(W2, w2b, DD * NHID);

  // 2) projection GEMM: q = x @ Wp + bp
  gemm_wmma_bf16<0><<<dim3(DD / 64, Mrows / A_ROWS), 256, 0, stream>>>(
      xbf, wptb, bp, nullptr, qf, nullptr, Mrows, DD, DD);

  // 3) sparse-mode DFT, 4) mixing, 5) synthesis + residual
  k_dft_modes<<<BB * DD, 256, 0, stream>>>(qf, midx, Xre, Xim);
  k_mix<<<(BB * HH * EE * MMM + 255) / 256, 256, 0, stream>>>(
      Xre, Xim, w_real, w_imag, Ore, Oim);
  k_synth_residual<<<(unsigned)(BTD / 256), 256, 0, stream>>>(
      x, Ore, Oim, midx, x1f, x1bf);

  // 6) FFN up: y = relu(x1 @ W1^T), bf16
  gemm_wmma_bf16<1><<<dim3(NHID / 64, Mrows / A_ROWS), 256, 0, stream>>>(
      x1bf, w1b, nullptr, nullptr, nullptr, ybf, Mrows, NHID, DD);

  // 7) FFN down + residual: out = x1 + y @ W2^T
  gemm_wmma_bf16<2><<<dim3(DD / 64, Mrows / A_ROWS), 256, 0, stream>>>(
      ybf, w2b, nullptr, x1f, out, nullptr, Mrows, DD, NHID);
}
